// CausalWanSelfAttention_24129126269481
// MI455X (gfx1250) — matrix operand order repaired and checked by reading, not compiled, not run on hardware
//
#include <hip/hip_runtime.h>
#include <hip/hip_bf16.h>

// ---------------------------------------------------------------------------
// CausalWanSelfAttention for MI455X (gfx1250, wave32, WMMA + TDM)
// B=1, S=3120 (=6*20*26), DIM=1536, NH=12, HD=128, frame=520 tokens
// ---------------------------------------------------------------------------

typedef __attribute__((ext_vector_type(16))) __bf16 v16bf;
typedef __attribute__((ext_vector_type(8)))  float  v8f;

#define S_TOK 3120
#define DIMN  1536
#define NHEAD 12
#define HDIM  128
#define HWTOK 520   // H*W tokens per frame
#define LOCF  4     // LOCAL_FRAMES
#define SINKF 1     // SINK_FRAMES

#if defined(__has_builtin)
#  if __has_builtin(__builtin_amdgcn_tensor_load_to_lds)
#    define HAVE_TDM 1
#  endif
#endif
#ifndef HAVE_TDM
#  define HAVE_TDM 0
#endif

__device__ __forceinline__ void wait_tensorcnt0() {
#if defined(__has_builtin) && __has_builtin(__builtin_amdgcn_s_wait_tensorcnt)
  __builtin_amdgcn_s_wait_tensorcnt(0);
#else
  asm volatile("s_wait_tensorcnt 0x0" ::: "memory");
#endif
}

#if HAVE_TDM
typedef unsigned tdm_u4 __attribute__((ext_vector_type(4)));
typedef int      tdm_i8 __attribute__((ext_vector_type(8)));
typedef int      tdm_i4 __attribute__((ext_vector_type(4)));

// 2-D tile load Global->LDS via Tensor Data Mover.
// dims/strides in elements (data_size = 2 bytes, bf16).
// pad_interval_code / pad_amount_code per D# group1 encodings.
__device__ __forceinline__ void tdm_load_2d(
    const void* gsrc, unsigned lds_byte_addr,
    int tensor_d0, int tensor_d1, int tile_d0, int tile_d1,
    long long row_stride, int pad_interval_code, int pad_amount_code) {
  unsigned long long ga = (unsigned long long)gsrc;
  tdm_u4 g0;
  g0[0] = 1u;                                          // count=1, user D#
  g0[1] = lds_byte_addr;                               // LDS dest
  g0[2] = (unsigned)(ga & 0xFFFFFFFFu);                // global addr lo
  g0[3] = (unsigned)((ga >> 32) & 0x1FFFFFFu) | (2u << 30);  // hi + type=2
  tdm_i8 g1;
  unsigned w0 = (1u << 16)                             // data_size = 2 bytes
              | (1u << 20)                             // pad_enable
              | ((unsigned)pad_interval_code << 22)
              | ((unsigned)pad_amount_code  << 25);
  g1[0] = (int)w0;
  g1[1] = (int)((unsigned)tensor_d0 << 16);            // dim0[15:0] @ bits63:48
  g1[2] = (int)((((unsigned)tensor_d0 >> 16) & 0xFFFFu) |
                ((unsigned)tensor_d1 << 16));
  g1[3] = (int)((((unsigned)tensor_d1 >> 16) & 0xFFFFu) |
                ((unsigned)tile_d0 << 16));
  g1[4] = (int)((unsigned)tile_d1 & 0xFFFFu);          // tile_dim1, tile_dim2=0
  g1[5] = (int)(unsigned)(row_stride & 0xFFFFFFFFull); // dim0_stride lo32
  g1[6] = (int)(unsigned)((row_stride >> 32) & 0xFFFFull);
  g1[7] = 0;
  tdm_i4 z4 = {};
#if defined(__clang_major__) && (__clang_major__ >= 23)
  tdm_i8 z8 = {};
  __builtin_amdgcn_tensor_load_to_lds(g0, g1, z4, z4, z8, 0);
#else
  __builtin_amdgcn_tensor_load_to_lds(g0, g1, z4, z4, 0);
#endif
}
#endif  // HAVE_TDM

__device__ __forceinline__ __bf16 f2bf(float f) {
  union { float f; unsigned u; } in;
  in.f = f;
  unsigned u = in.u;
  u += 0x7FFFu + ((u >> 16) & 1u);          // round to nearest even
  union { unsigned short s; __bf16 b; } out;
  out.s = (unsigned short)(u >> 16);
  return out.b;
}

// ---------------- elementwise converts ----------------

__global__ __launch_bounds__(256) void cvt_f32_bf16(
    const float* __restrict__ in, __bf16* __restrict__ out, int n) {
  int i = blockIdx.x * 256 + threadIdx.x;
  if (i < n) out[i] = f2bf(in[i]);
}

// WT[n*dim + k] = bf16( W[k*dim + n] )
__global__ __launch_bounds__(256) void transpose_f32_bf16(
    const float* __restrict__ W, __bf16* __restrict__ WT, int dim) {
  int i = blockIdx.x * 256 + threadIdx.x;
  if (i < dim * dim) {
    int n = i / dim, k = i % dim;
    WT[i] = f2bf(W[(size_t)k * dim + n]);
  }
}

// ---------------- GEMM: C(MxN) = A(MxK) * W(KxN) + bias, K=N=1536 ----------
// A bf16 row-major, BT = W^T bf16 row-major ([n][k]), C f32.
// Block tile 64(M) x 128(N), 8 waves, K-step 32, TDM double-buffered LDS.

__global__ __launch_bounds__(256) void gemm_xw_bf16(
    const __bf16* __restrict__ A,
    const __bf16* __restrict__ BT,
    const float*  __restrict__ bias,
    float* __restrict__ C, int M) {
  __shared__ __attribute__((aligned(16))) __bf16 As[2][64 * 40];
  __shared__ __attribute__((aligned(16))) __bf16 Bs[2][128 * 40];

  const int tid  = threadIdx.x;
  const int w    = tid >> 5;
  const int lane = tid & 31;
  const int lh   = lane >> 4;
  const int ln   = lane & 15;
  const int mbase = blockIdx.x * 64;
  const int nbase = blockIdx.y * 128;
  const int mt = w >> 1;          // 0..3 : 16-row tile
  const int nh = w & 1;           // 0..1 : 64-col half

  v8f zero = {};
  v8f acc[4];
#pragma unroll
  for (int j = 0; j < 4; ++j) acc[j] = zero;

  auto mma_step = [&](const __bf16* As_, const __bf16* Bs_) {
    v16bf af;
#pragma unroll
    for (int e = 0; e < 16; ++e) {
      int kk = e + (e & 8) + (lh << 3);          // A 16x32 layout
      af[e] = As_[(mt * 16 + ln) * 40 + kk];
    }
#pragma unroll
    for (int j = 0; j < 4; ++j) {
      v16bf bfr;
#pragma unroll
      for (int e = 0; e < 16; ++e) {
        int kk = e + (e & 8) + (lh << 3);        // B 32x16 layout
        bfr[e] = Bs_[(nh * 64 + j * 16 + ln) * 40 + kk];
      }
      acc[j] = __builtin_amdgcn_wmma_f32_16x16x32_bf16(
          false, af, false, bfr, (short)0, acc[j], false, false);
    }
  };

#if HAVE_TDM
  int rowsA = M - mbase; if (rowsA > 64) rowsA = 64;
  auto issueAB = [&](int kt, int b) {
    int k0 = kt * 32;
    // 32-elem rows (64B = 16 DW -> interval code 3), pad 16B = 4 DW -> code 3
    tdm_load_2d(A + (size_t)mbase * DIMN + k0,
                (unsigned)(size_t)(void*)&As[b][0],
                32, rowsA, 32, 64, (long long)DIMN, 3, 3);
    tdm_load_2d(BT + (size_t)nbase * DIMN + k0,
                (unsigned)(size_t)(void*)&Bs[b][0],
                32, 128, 32, 128, (long long)DIMN, 3, 3);
  };
  if (w == 0) { issueAB(0, 0); wait_tensorcnt0(); }
  __syncthreads();
  for (int kt = 0; kt < DIMN / 32; ++kt) {
    int cur = kt & 1;
    if (w == 0 && (kt + 1) < DIMN / 32) issueAB(kt + 1, cur ^ 1);
    mma_step(As[cur], Bs[cur]);
    if (w == 0) wait_tensorcnt0();
    __syncthreads();
  }
#else
  const int arow = (tid * 8) >> 5;
  const int acol = (tid * 8) & 31;
  int ga = mbase + arow; if (ga >= M) ga = M - 1;
  const int brow = (tid * 16) >> 5;
  const int bcol = (tid * 16) & 31;
  for (int k0 = 0; k0 < DIMN; k0 += 32) {
    *(float4*)&As[0][arow * 40 + acol] =
        *(const float4*)&A[(size_t)ga * DIMN + k0 + acol];
    *(float4*)&Bs[0][brow * 40 + bcol] =
        *(const float4*)&BT[(size_t)(nbase + brow) * DIMN + k0 + bcol];
    *(float4*)&Bs[0][brow * 40 + bcol + 8] =
        *(const float4*)&BT[(size_t)(nbase + brow) * DIMN + k0 + bcol + 8];
    __syncthreads();
    mma_step(As[0], Bs[0]);
    __syncthreads();
  }
#endif

  // epilogue: bias hoisted per column; fast path for full M tiles
  const bool full = (mbase + 64 <= M);
#pragma unroll
  for (int j = 0; j < 4; ++j) {
    int col = nbase + nh * 64 + j * 16 + ln;
    float bc = bias[col];
    if (full) {
      float* cp = C + (size_t)(mbase + mt * 16 + (lh << 3)) * DIMN + col;
#pragma unroll
      for (int r = 0; r < 8; ++r) cp[(size_t)r * DIMN] = acc[j][r] + bc;
    } else {
#pragma unroll
      for (int r = 0; r < 8; ++r) {
        int row = mbase + mt * 16 + r + (lh << 3);
        if (row < M) C[(size_t)row * DIMN + col] = acc[j][r] + bc;
      }
    }
  }
}

// ---------------- RMSNorm + 3D RoPE -> bf16 ----------------

__global__ __launch_bounds__(256) void rmsnorm_rope(
    const float* __restrict__ pre, const float* __restrict__ g,
    const float* __restrict__ fc, const float* __restrict__ fs,
    __bf16* __restrict__ out) {
  int s = blockIdx.x;
  int t = threadIdx.x;
  const float* row = pre + (size_t)s * DIMN;

  float ss = 0.f;
#pragma unroll
  for (int i = 0; i < 6; ++i) {
    float v = row[t + 256 * i];
    ss += v * v;
  }
  for (int m = 1; m < 32; m <<= 1) ss += __shfl_xor(ss, m);
  __shared__ float red[8];
  if ((t & 31) == 0) red[t >> 5] = ss;
  __syncthreads();
  float tot = 0.f;
#pragma unroll
  for (int i = 0; i < 8; ++i) tot += red[i];
  float rstd = rsqrtf(tot * (1.0f / (float)DIMN) + 1e-6f);

  int f  = s / HWTOK;
  int hh = (s / 26) % 20;
  int ww = s % 26;
#pragma unroll
  for (int i = 0; i < 3; ++i) {
    int p = t + 256 * i;              // pair index 0..767
    int c = p & 63;                   // channel within head (C=64)
    int prow = (c < 22) ? f : (c < 43) ? hh : ww;
    float cv = fc[prow * 64 + c];
    float sv = fs[prow * 64 + c];
    float xr = row[2 * p]     * rstd * g[2 * p];
    float xi = row[2 * p + 1] * rstd * g[2 * p + 1];
    out[(size_t)s * DIMN + 2 * p]     = f2bf(xr * cv - xi * sv);
    out[(size_t)s * DIMN + 2 * p + 1] = f2bf(xr * sv + xi * cv);
  }
}

// ---------------- Flash attention ----------------
// block = (q-block of 128 rows, head); 8 waves, wave = 16 q rows.
// K/V streamed 32 keys/iter via TDM (double-buffered); WMMA bf16.

__global__ __launch_bounds__(256) void flash_attn(
    const __bf16* __restrict__ Q, const __bf16* __restrict__ K,
    const __bf16* __restrict__ V, __bf16* __restrict__ O) {
  __shared__ __attribute__((aligned(16))) __bf16 KtB[2][32 * 136];
  __shared__ __attribute__((aligned(16))) __bf16 VtB[2][32 * 136];
  __shared__ __attribute__((aligned(16))) __bf16 Pb[8 * 16 * 32];

  const int head  = blockIdx.y;
  const int qbase = blockIdx.x * 128;
  const int w     = threadIdx.x >> 5;
  const int lane  = threadIdx.x & 31;
  const int lh    = lane >> 4;
  const int ln    = lane & 15;
  const int qr0   = qbase + w * 16;

  // preload Q fragments (16 rows x 128 head-dim = 4 A-fragments)
  v16bf qf[4];
  {
    int qrow = qr0 + ln; if (qrow >= S_TOK) qrow = S_TOK - 1;
    const __bf16* qp = Q + (size_t)qrow * DIMN + head * HDIM;
#pragma unroll
    for (int c = 0; c < 4; ++c) {
#pragma unroll
      for (int e = 0; e < 16; ++e) {
        int kd = 32 * c + e + (e & 8) + (lh << 3);
        qf[c][e] = qp[kd];
      }
    }
  }

  v8f zero = {};
  v8f acc[8];
#pragma unroll
  for (int t = 0; t < 8; ++t) acc[t] = zero;
  float mrow[8], lrow[8];
#pragma unroll
  for (int r = 0; r < 8; ++r) { mrow[r] = -1e30f; lrow[r] = 0.f; }

  const int fi_lo = ((qbase < S_TOK) ? qbase : (S_TOK - 1)) / HWTOK;
  int qtop = qbase + 127; if (qtop >= S_TOK) qtop = S_TOK - 1;
  const int fi_hi = qtop / HWTOK;
  const float scale = 0.08838834764831845f;   // 1/sqrt(128)

  auto tile_ok = [&](int kb) -> bool {   // safe superset of "any allowed"
    int fj_lo = kb / HWTOK;
    int kbt = kb + 31; if (kbt >= S_TOK) kbt = S_TOK - 1;
    int fj_hi = kbt / HWTOK;
    return (fj_lo <= fi_hi) && (((fi_lo - fj_hi) < LOCF) || (fj_lo < SINKF));
  };

  auto do_tile = [&](int kb, const __bf16* Kt, const __bf16* Vt) {
    // scores: two 16x16 tiles (keys kb..+15, kb+16..+31), K-dim = 128
    v8f sc[2];
#pragma unroll
    for (int t2 = 0; t2 < 2; ++t2) {
      sc[t2] = zero;
#pragma unroll
      for (int c = 0; c < 4; ++c) {
        v16bf bfr;
#pragma unroll
        for (int e = 0; e < 16; ++e) {
          int kd = 32 * c + e + (e & 8) + (lh << 3);
          bfr[e] = Kt[(t2 * 16 + ln) * 136 + kd];
        }
        sc[t2] = __builtin_amdgcn_wmma_f32_16x16x32_bf16(
            false, qf[c], false, bfr, (short)0, sc[t2], false, false);
      }
    }

    // mask + online softmax
    float pv[2][8];
    float corr[8];
#pragma unroll
    for (int r = 0; r < 8; ++r) {
      int qpos = qr0 + r + (lh << 3);
      int qc = (qpos < S_TOK) ? qpos : (S_TOK - 1);
      int fi = qc / HWTOK;
      float v2[2];
      float rmax = -1e30f;
#pragma unroll
      for (int t2 = 0; t2 < 2; ++t2) {
        int kpos = kb + t2 * 16 + ln;
        int fj = kpos / HWTOK;
        bool ok = (kpos < S_TOK) && (fj <= fi) &&
                  (((fi - fj) < LOCF) || (fj < SINKF));
        float s = sc[t2][r] * scale;
        s = ok ? s : -1e30f;
        v2[t2] = s;
        rmax = fmaxf(rmax, s);
      }
#pragma unroll
      for (int m = 1; m < 16; m <<= 1) rmax = fmaxf(rmax, __shfl_xor(rmax, m));
      float mnew = fmaxf(mrow[r], rmax);
      float cf = expf(mrow[r] - mnew);
      float psum = 0.f;
#pragma unroll
      for (int t2 = 0; t2 < 2; ++t2) {
        float p = (v2[t2] > -1e29f) ? expf(v2[t2] - mnew) : 0.f;
        pv[t2][r] = p;
        psum += p;
      }
#pragma unroll
      for (int m = 1; m < 16; m <<= 1) psum += __shfl_xor(psum, m);
      lrow[r] = lrow[r] * cf + psum;
      mrow[r] = mnew;
      corr[r] = cf;
    }

#pragma unroll
    for (int t = 0; t < 8; ++t)
#pragma unroll
      for (int r = 0; r < 8; ++r) acc[t][r] *= corr[r];

    // P (C-layout) -> per-wave LDS scratch -> A-fragment layout
    __bf16* pb = &Pb[w * 512];
#pragma unroll
    for (int t2 = 0; t2 < 2; ++t2)
#pragma unroll
      for (int r = 0; r < 8; ++r)
        pb[(r + (lh << 3)) * 32 + t2 * 16 + ln] = f2bf(pv[t2][r]);
    asm volatile("s_wait_dscnt 0" ::: "memory");
    v16bf af;
#pragma unroll
    for (int e = 0; e < 16; ++e) {
      int kk = e + (e & 8) + (lh << 3);
      af[e] = pb[ln * 32 + kk];
    }

    // O += P(16x32) * V(32x128)
#pragma unroll
    for (int t = 0; t < 8; ++t) {
      v16bf vfr;
#pragma unroll
      for (int e = 0; e < 16; ++e) {
        int kk = e + (e & 8) + (lh << 3);
        vfr[e] = Vt[kk * 136 + t * 16 + ln];
      }
      acc[t] = __builtin_amdgcn_wmma_f32_16x16x32_bf16(
          false, af, false, vfr, (short)0, acc[t], false, false);
    }
  };

#if HAVE_TDM
  auto issueKV = [&](int kb, int b) {
    int rows = S_TOK - kb; if (rows > 32) rows = 32;   // OOB rows zero-filled
    // 128-elem rows (256B = 64 DW -> interval code 5), pad 16B = 4 DW -> code 3
    tdm_load_2d(K + (size_t)kb * DIMN + head * HDIM,
                (unsigned)(size_t)(void*)&KtB[b][0],
                128, rows, 128, 32, (long long)DIMN, 5, 3);
    tdm_load_2d(V + (size_t)kb * DIMN + head * HDIM,
                (unsigned)(size_t)(void*)&VtB[b][0],
                128, rows, 128, 32, (long long)DIMN, 5, 3);
  };
  int kb = 0, cur = 0;                 // tile 0 is always allowed (sink)
  if (w == 0) { issueKV(0, 0); wait_tensorcnt0(); }
  __syncthreads();
  while (kb < S_TOK) {
    int knext = kb + 32;
    while (knext < S_TOK && !tile_ok(knext)) knext += 32;
    if (w == 0 && knext < S_TOK) issueKV(knext, cur ^ 1);
    do_tile(kb, KtB[cur], VtB[cur]);
    if (w == 0) wait_tensorcnt0();
    __syncthreads();
    kb = knext; cur ^= 1;
  }
#else
  const int rr = threadIdx.x >> 3;
  const int cc = (threadIdx.x & 7) * 16;
  for (int kb = 0; kb < S_TOK; kb += 32) {
    if (!tile_ok(kb)) continue;
    {
      int krow = kb + rr; if (krow >= S_TOK) krow = S_TOK - 1;
      const float4* sk = (const float4*)(K + (size_t)krow * DIMN + head * HDIM + cc);
      const float4* sv = (const float4*)(V + (size_t)krow * DIMN + head * HDIM + cc);
      float4* dk = (float4*)&KtB[0][rr * 136 + cc];
      float4* dv = (float4*)&VtB[0][rr * 136 + cc];
      dk[0] = sk[0]; dk[1] = sk[1];
      dv[0] = sv[0]; dv[1] = sv[1];
      if (kb + 32 < S_TOK) {
        __builtin_prefetch(K + (size_t)(kb + 32 + rr) * DIMN + head * HDIM + cc, 0, 1);
        __builtin_prefetch(V + (size_t)(kb + 32 + rr) * DIMN + head * HDIM + cc, 0, 1);
      }
    }
    __syncthreads();
    do_tile(kb, KtB[0], VtB[0]);
    __syncthreads();
  }
#endif

  // normalize + store bf16
#pragma unroll
  for (int r = 0; r < 8; ++r) {
    int qpos = qr0 + r + (lh << 3);
    if (qpos < S_TOK) {
      float inv = (lrow[r] > 0.f) ? (1.0f / lrow[r]) : 0.f;
#pragma unroll
      for (int t = 0; t < 8; ++t)
        O[(size_t)qpos * DIMN + head * HDIM + t * 16 + ln] = f2bf(acc[t][r] * inv);
    }
  }
}

// ---------------- host launcher ----------------

extern "C" void kernel_launch(void* const* d_in, const int* in_sizes, int n_in,
                              void* d_out, int out_size, void* d_ws, size_t ws_size,
                              hipStream_t stream) {
  (void)in_sizes; (void)n_in; (void)out_size; (void)ws_size;
  const float* x  = (const float*)d_in[0];
  const float* fc = (const float*)d_in[3];
  const float* fs = (const float*)d_in[4];
  const float* Wq = (const float*)d_in[5];  const float* bq = (const float*)d_in[6];
  const float* Wk = (const float*)d_in[7];  const float* bk = (const float*)d_in[8];
  const float* Wv = (const float*)d_in[9];  const float* bv = (const float*)d_in[10];
  const float* Wo = (const float*)d_in[11]; const float* bo = (const float*)d_in[12];
  const float* gq = (const float*)d_in[13]; const float* gk = (const float*)d_in[14];

  char* ws = (char*)d_ws;
  size_t off = 0;
  auto take = [&](size_t bytes) -> char* {
    char* p = ws + off;
    off = (off + bytes + 255) & ~(size_t)255;
    return p;
  };
  __bf16* xb  = (__bf16*)take((size_t)S_TOK * DIMN * 2);
  __bf16* WTq = (__bf16*)take((size_t)DIMN * DIMN * 2);
  __bf16* WTk = (__bf16*)take((size_t)DIMN * DIMN * 2);
  __bf16* WTv = (__bf16*)take((size_t)DIMN * DIMN * 2);
  __bf16* WTo = (__bf16*)take((size_t)DIMN * DIMN * 2);
  float*  pre = (float*) take((size_t)S_TOK * DIMN * 4);
  __bf16* qb  = (__bf16*)take((size_t)S_TOK * DIMN * 2);
  __bf16* kb2 = (__bf16*)take((size_t)S_TOK * DIMN * 2);
  __bf16* vb  = (__bf16*)take((size_t)S_TOK * DIMN * 2);
  __bf16* ob  = (__bf16*)pre;  // alias: pre is dead once vb is produced

  const int nX = S_TOK * DIMN;
  const int nW = DIMN * DIMN;
  cvt_f32_bf16<<<(nX + 255) / 256, 256, 0, stream>>>(x, xb, nX);
  transpose_f32_bf16<<<(nW + 255) / 256, 256, 0, stream>>>(Wq, WTq, DIMN);
  transpose_f32_bf16<<<(nW + 255) / 256, 256, 0, stream>>>(Wk, WTk, DIMN);
  transpose_f32_bf16<<<(nW + 255) / 256, 256, 0, stream>>>(Wv, WTv, DIMN);
  transpose_f32_bf16<<<(nW + 255) / 256, 256, 0, stream>>>(Wo, WTo, DIMN);

  dim3 gg((S_TOK + 63) / 64, DIMN / 128);
  gemm_xw_bf16<<<gg, 256, 0, stream>>>(xb, WTq, bq, pre, S_TOK);
  rmsnorm_rope<<<S_TOK, 256, 0, stream>>>(pre, gq, fc, fs, qb);
  gemm_xw_bf16<<<gg, 256, 0, stream>>>(xb, WTk, bk, pre, S_TOK);
  rmsnorm_rope<<<S_TOK, 256, 0, stream>>>(pre, gk, fc, fs, kb2);
  gemm_xw_bf16<<<gg, 256, 0, stream>>>(xb, WTv, bv, pre, S_TOK);
  cvt_f32_bf16<<<(nX + 255) / 256, 256, 0, stream>>>(pre, vb, nX);

  dim3 fg((S_TOK + 127) / 128, NHEAD);
  flash_attn<<<fg, 256, 0, stream>>>(qb, kb2, vb, ob);

  gemm_xw_bf16<<<gg, 256, 0, stream>>>(ob, WTo, bo, (float*)d_out, S_TOK);
}
